// WindowAttention_54932631716191
// MI455X (gfx1250) — matrix operand order repaired and checked by reading, not compiled, not run on hardware
//
#include <hip/hip_runtime.h>

typedef __attribute__((ext_vector_type(16))) __bf16 v16bf;
typedef __attribute__((ext_vector_type(4)))  __bf16 v4bf;
typedef __attribute__((ext_vector_type(2)))  __bf16 v2bf;
typedef __attribute__((ext_vector_type(8)))  float  v8f;

#define DIMC  128
#define NTOK  49
#define HEADS 4
#define HD    32
#define NWIN  64

// LDS strides (elements)
#define XB_STR 136   // x tile / O tile rows (bf16), 64 rows
#define QK_STR 40    // q/k rows (bf16)
#define VT_STR 72    // v-transposed rows (bf16)
#define S_STR  68    // S rows (fp32); bf16 P overlays row start

// ---- WMMA fragment loaders (16x16x32 bf16, wave32 layouts per CDNA5 ISA) ----

// A (MxK): lane&15 = row; lane>=16 handles K+8 half. VGPR0..3 = K[kb..kb+7], VGPR4..7 = K[kb+16..kb+23]
__device__ __forceinline__ v16bf frag_a(const __bf16* base, int row0, int col0, int stride, int lane) {
  union { v16bf v; unsigned int u[8]; } r;
  int row = row0 + (lane & 15);
  int kb  = col0 + ((lane >> 4) << 3);
  const unsigned int* p0 = (const unsigned int*)(base + row * stride + kb);
  const unsigned int* p1 = (const unsigned int*)(base + row * stride + kb + 16);
#pragma unroll
  for (int i = 0; i < 4; i++) r.u[i] = p0[i];
#pragma unroll
  for (int i = 0; i < 4; i++) r.u[4 + i] = p1[i];
  return r.v;
}

// B (KxN) supplied as row-major [N][K]: lane&15 = column n; lanes>=16 handle K+8 half. VGPR0..3 = K[kb..kb+7], VGPR4..7 = K[kb+16..kb+23]
__device__ __forceinline__ v16bf frag_b(const __bf16* base, int n0, int k0, int stride, int lane) {
  union { v16bf v; unsigned int u[8]; } r;
  int n  = n0 + (lane & 15);
  int kb = k0 + ((lane >> 4) << 3);
  const unsigned int* p = (const unsigned int*)(base + n * stride + kb);
#pragma unroll
  for (int i = 0; i < 8; i++) r.u[i] = p[(i & 3) + ((i >> 2) << 3)];
  return r.v;
}

// ---- Prep: weights fp32->bf16, gather relative-position bias into [4][64][64] fp32 ----
__global__ void winattn_prep(const float* __restrict__ qkv_w, const float* __restrict__ proj_w,
                             const float* __restrict__ bias_table, const int* __restrict__ rel_index,
                             __bf16* __restrict__ qkvw_bf, __bf16* __restrict__ projw_bf,
                             float* __restrict__ biasg) {
  int gid = blockIdx.x * blockDim.x + threadIdx.x;
  int str = gridDim.x * blockDim.x;
  for (int i = gid; i < 3 * DIMC * DIMC; i += str) qkvw_bf[i] = (__bf16)qkv_w[i];
  for (int i = gid; i < DIMC * DIMC; i += str)     projw_bf[i] = (__bf16)proj_w[i];
  for (int i = gid; i < HEADS * 64 * 64; i += str) {
    int h = i >> 12, rem = i & 4095, m = rem >> 6, n = rem & 63;
    float v = 0.0f;
    if (m < NTOK && n < NTOK) v = bias_table[rel_index[m * NTOK + n] * HEADS + h];
    biasg[i] = v;
  }
}

// ---- Fused window attention: one workgroup per window ----
__global__ __launch_bounds__(256)
void winattn_fused(const float* __restrict__ x, const float* __restrict__ mask,
                   const float* __restrict__ qkv_b, const float* __restrict__ proj_b,
                   const __bf16* __restrict__ qkvw, const __bf16* __restrict__ projw,
                   const float* __restrict__ biasg, float* __restrict__ out) {
  __shared__ __bf16 xb[64 * XB_STR];              // x tile, reused for O tile
  __shared__ __bf16 qs[HEADS * 64 * QK_STR];
  __shared__ __bf16 kk_[HEADS * 64 * QK_STR];
  __shared__ __bf16 vT[HEADS * HD * VT_STR];
  __shared__ float  Ss[HEADS * 64 * S_STR];       // S fp32; bf16 P overlays row starts

  const int tid  = threadIdx.x;
  const int lane = tid & 31;
  // wave id is uniform across the wave: pin it (and thus all tile indices /
  // kind / h below) into SGPRs so tile loops and q/k/v selection use scalar
  // branches instead of exec-mask divergence.
  const int wid  = __builtin_amdgcn_readfirstlane(tid >> 5);
  const int b    = blockIdx.x;
  const float scale = 0.17677669529663687f;       // 32^-0.5

  // Stage 0: zero padding rows 49..63, load x (float4) -> bf16 quads
  for (int i = tid; i < 15 * DIMC; i += 256) {
    int m = 49 + (i >> 7), c = i & 127;
    xb[m * XB_STR + c] = (__bf16)0.0f;
  }
  {
    const float4* xp4 = (const float4*)(x + (size_t)b * NTOK * DIMC);
    for (int i = tid; i < NTOK * (DIMC / 4); i += 256) {   // 1568 quads
      int m = i >> 5, c4 = (i & 31) << 2;
      float4 f = xp4[i];
      v4bf q;
      q.x = (__bf16)f.x; q.y = (__bf16)f.y; q.z = (__bf16)f.z; q.w = (__bf16)f.w;
      *(v4bf*)&xb[m * XB_STR + c4] = q;
    }
  }
  __syncthreads();

  // Stage 1: QKV GEMM  [64x128]x[128x384]; scatter q (scaled), k, v^T to LDS.
  // kind/h derived from scalar nt -> s_cbranch, full EXEC everywhere.
  for (int t = wid; t < 96; t += 8) {
    int mt = t & 3, nt = t >> 2;
    v8f acc = {};
#pragma unroll
    for (int kk = 0; kk < 4; kk++) {
      v16bf a  = frag_a(xb, mt * 16, kk * 32, XB_STR, lane);
      v16bf bm = frag_b(qkvw, nt * 16, kk * 32, DIMC, lane);
      acc = __builtin_amdgcn_wmma_f32_16x16x32_bf16(false, a, false, bm, (short)0, acc, false, false);
    }
    int kind = nt >> 3;                 // 0=q(nt 0-7) 1=k(8-15) 2=v(16-23)  [scalar]
    int h    = (nt >> 1) & 3;           // [scalar]
    int d    = ((nt & 1) << 4) + (lane & 15);
    float bias = qkv_b[nt * 16 + (lane & 15)];
    int mbase = mt * 16 + ((lane >> 4) << 3);
    if (kind == 0) {
#pragma unroll
      for (int r = 0; r < 8; r++)
        qs[(h * 64 + mbase + r) * QK_STR + d] = (__bf16)((acc[r] + bias) * scale);
    } else if (kind == 1) {
#pragma unroll
      for (int r = 0; r < 8; r++)
        kk_[(h * 64 + mbase + r) * QK_STR + d] = (__bf16)(acc[r] + bias);
    } else {
      // consecutive m at fixed d row of vT -> packs into one ds_store_b128
      v2bf* vp = (v2bf*)&vT[(h * HD + d) * VT_STR + mbase];
#pragma unroll
      for (int rr = 0; rr < 4; rr++) {
        v2bf pv;
        pv.x = (__bf16)(acc[2 * rr] + bias);
        pv.y = (__bf16)(acc[2 * rr + 1] + bias);
        vp[rr] = pv;
      }
    }
  }
  __syncthreads();

  // Stage 2: S = q @ k^T per head (K = hd = 32 -> single WMMA per tile)
  for (int t = wid; t < 64; t += 8) {
    int h = t >> 4, mt = (t >> 2) & 3, nt = t & 3;
    v16bf a  = frag_a(qs + h * 64 * QK_STR, mt * 16, 0, QK_STR, lane);
    v16bf bm = frag_b(kk_ + h * 64 * QK_STR, nt * 16, 0, QK_STR, lane);
    v8f acc = {};
    acc = __builtin_amdgcn_wmma_f32_16x16x32_bf16(false, a, false, bm, (short)0, acc, false, false);
    int n = nt * 16 + (lane & 15);
    int mbase = mt * 16 + ((lane >> 4) << 3);
#pragma unroll
    for (int r = 0; r < 8; r++)
      Ss[(h * 64 + mbase + r) * S_STR + n] = acc[r];
  }
  __syncthreads();

  // Stage 3: bias + mask + softmax; one row per thread (256 rows exactly).
  // Pack bf16 probabilities in place over the fp32 row (write offset trails read offset).
  {
    int h = tid >> 6, m = tid & 63;
    float* Srow = &Ss[(h * 64 + m) * S_STR];
    const float* bi = &biasg[(h * 64 + m) * 64];
    int w = b & (NWIN - 1);
    const float* mrow = mask + ((size_t)w * NTOK + m) * NTOK;   // only read when m<49
    bool vm = (m < NTOK);
    float mx = -1e30f;
    for (int n = 0; n < 64; n++) {
      float s;
      if (n < NTOK) s = Srow[n] + bi[n] + (vm ? mrow[n] : 0.0f);
      else          s = -1e9f;
      Srow[n] = s;
      mx = fmaxf(mx, s);
    }
    float sum = 0.0f;
    for (int n = 0; n < 64; n++) {
      float e = __expf(Srow[n] - mx);
      Srow[n] = e;
      sum += e;
    }
    float inv = 1.0f / sum;
    v2bf* prow = (v2bf*)Srow;
    for (int i = 0; i < 32; i++) {
      v2bf pv;
      pv.x = (__bf16)(Srow[2 * i] * inv);
      pv.y = (__bf16)(Srow[2 * i + 1] * inv);
      prow[i] = pv;
    }
  }
  __syncthreads();

  // Stage 4: O = P @ v  (K = 64 over tokens), write O bf16 into xb (x tile is dead)
  for (int t = wid; t < 32; t += 8) {
    int h = t >> 3, mt = (t >> 1) & 3, nt = t & 1;
    v8f acc = {};
#pragma unroll
    for (int kk = 0; kk < 2; kk++) {
      v16bf a  = frag_a((const __bf16*)Ss, h * 64 + mt * 16, kk * 32, S_STR * 2, lane);
      v16bf bm = frag_b(vT + h * HD * VT_STR, nt * 16, kk * 32, VT_STR, lane);
      acc = __builtin_amdgcn_wmma_f32_16x16x32_bf16(false, a, false, bm, (short)0, acc, false, false);
    }
    int d = nt * 16 + (lane & 15);
    int mbase = mt * 16 + ((lane >> 4) << 3);
#pragma unroll
    for (int r = 0; r < 8; r++)
      xb[(mbase + r) * XB_STR + h * HD + d] = (__bf16)acc[r];
  }
  __syncthreads();

  // Stage 5: out = O @ proj_w^T + proj_b, fp32 stores for valid rows (coalesced across lanes)
  float* outp = out + (size_t)b * NTOK * DIMC;
  for (int t = wid; t < 32; t += 8) {
    int mt = t & 3, nt = t >> 2;
    v8f acc = {};
#pragma unroll
    for (int kk = 0; kk < 4; kk++) {
      v16bf a  = frag_a(xb, mt * 16, kk * 32, XB_STR, lane);
      v16bf bm = frag_b(projw, nt * 16, kk * 32, DIMC, lane);
      acc = __builtin_amdgcn_wmma_f32_16x16x32_bf16(false, a, false, bm, (short)0, acc, false, false);
    }
    int c = nt * 16 + (lane & 15);
    float pb = proj_b[c];
    int mbase = mt * 16 + ((lane >> 4) << 3);
#pragma unroll
    for (int r = 0; r < 8; r++) {
      int m = mbase + r;
      if (m < NTOK) outp[m * DIMC + c] = acc[r] + pb;
    }
  }
}

extern "C" void kernel_launch(void* const* d_in, const int* in_sizes, int n_in,
                              void* d_out, int out_size, void* d_ws, size_t ws_size,
                              hipStream_t stream) {
  const float* x          = (const float*)d_in[0];
  const float* mask       = (const float*)d_in[1];
  const float* qkv_w      = (const float*)d_in[2];
  const float* qkv_b      = (const float*)d_in[3];
  const float* proj_w     = (const float*)d_in[4];
  const float* proj_b     = (const float*)d_in[5];
  const float* bias_table = (const float*)d_in[6];
  const int*   rel_index  = (const int*)d_in[7];

  char* ws = (char*)d_ws;
  __bf16* qkvw_bf = (__bf16*)ws;                        // 3*128*128*2 = 98304 B
  __bf16* projw_bf = (__bf16*)(ws + 98304);             // 128*128*2   = 32768 B
  float*  biasg    = (float*)(ws + 98304 + 32768);      // 4*64*64*4   = 65536 B

  winattn_prep<<<64, 256, 0, stream>>>(qkv_w, proj_w, bias_table, rel_index,
                                       qkvw_bf, projw_bf, biasg);
  winattn_fused<<<4096, 256, 0, stream>>>(x, mask, qkv_b, proj_b,
                                          qkvw_bf, projw_bf, biasg, (float*)d_out);
}
